// MoELayer_19825569038533
// MI455X (gfx1250) — matrix operand order repaired and checked by reading, not compiled, not run on hardware
//
#include <hip/hip_runtime.h>

// MoE FFN (proportional router): out = (relu(x@W1+b1)@W2 + b2) * exp(min(T, ln100))
// N=524288 tokens, D=128, H=512, E=8. Fused two-GEMM per 128-token tile,
// bf16 WMMA operands with fp32 accumulation on gfx1250 (CDNA5, wave32).
// Streaming tensors (x, out) use non-temporal hints so the 4MB expert weights
// stay resident in the 192MB L2 while 536MB streams through.

#define TOK_TILE 128
#define D_DIM    128
#define H_DIM    512
#define HC       128   // H processed per chunk
#define SA       136   // LDS row stride in bf16 elements (272B, 16B aligned)

typedef __attribute__((ext_vector_type(16))) __bf16 v16bf;
typedef __attribute__((ext_vector_type(8)))  float  v8f;
typedef __attribute__((ext_vector_type(4)))  float  fl4;

union Frag { v16bf v; uint4 u[2]; };

// round-to-nearest-even fp32 -> bf16
static __device__ inline unsigned short f2bf(float x) {
    unsigned u = __float_as_uint(x);
    u += 0x7FFFu + ((u >> 16) & 1u);
    return (unsigned short)(u >> 16);
}

// A-matrix fragment (16x32 bf16), row-major LDS tile [row][k], stride SA.
// ISA layout: lane L -> M=L%16; half=L/16; VGPR 0-3: K = 2g + 8*half,
// VGPR 4-7: K = 16 + 2(g-4) + 8*half  ==> two contiguous 16B reads.
static __device__ inline v16bf ldsA(const unsigned short* base, int row, int kk, int half) {
    Frag f;
    const unsigned short* r = base + row * SA + kk + 8 * half;
    f.u[0] = *(const uint4*)(r);
    f.u[1] = *(const uint4*)(r + 16);
    return f.v;
}

// B-matrix fragment (32x16 bf16) from LDS tile stored transposed [n][k].
// ISA layout: lane L -> N=L%16; half=L/16; VGPR 0-3: K = 2g + 8*half,
// VGPR 4-7: K = 16 + 2(g-4) + 8*half  ==> two contiguous 16B reads.
static __device__ inline v16bf ldsB(const unsigned short* base, int row, int kk, int half) {
    Frag f;
    const unsigned short* r = base + row * SA + kk + 8 * half;
    f.u[0] = *(const uint4*)(r);
    f.u[1] = *(const uint4*)(r + 16);
    return f.v;
}

static __device__ inline v8f wmma_bf16(v16bf a, v16bf b, v8f c) {
    return __builtin_amdgcn_wmma_f32_16x16x32_bf16(
        /*neg_a=*/false, a, /*neg_b=*/false, b,
        /*c_mod=*/(short)0, c, /*reuse_a=*/false, /*reuse_b=*/false);
}

__global__ __launch_bounds__(256) void moe_ffn_kernel(
    const float* __restrict__ x,
    const float* __restrict__ W1,
    const float* __restrict__ b1,
    const float* __restrict__ W2,
    const float* __restrict__ b2,
    const float* __restrict__ temperature,
    float* __restrict__ out)
{
    __shared__ __align__(16) unsigned short sX [TOK_TILE * SA];  // x tile, bf16, [tok][d]
    __shared__ __align__(16) unsigned short sW1[HC * SA];        // W1 chunk, transposed [h][d]
    __shared__ __align__(16) unsigned short sH [TOK_TILE * SA];  // relu(h) tile, bf16, [tok][h]
    __shared__ __align__(16) unsigned short sW2[D_DIM * SA];     // W2 chunk, transposed [d][h]

    const int tid  = threadIdx.x;
    const int lane = tid & 31;
    const int wave = tid >> 5;
    const int lm   = lane & 15;
    const int half = lane >> 4;
    const int m0   = (wave & 3) * 32;   // wave's token-row origin within tile
    const int n0   = (wave >> 2) * 64;  // wave's col origin (h or d)

    const long tok0 = (long)blockIdx.x * TOK_TILE;
    const int  e    = (int)(tok0 >> 16);           // 65536 tokens per expert
    const float* W1e = W1 + (size_t)e * D_DIM * H_DIM;
    const float* W2e = W2 + (size_t)e * H_DIM * D_DIM;
    const float* b1e = b1 + e * H_DIM;
    const float* b2e = b2 + e * D_DIM;

    // ---- stage x tile: fp32 global (b128, non-temporal) -> bf16 LDS ----
    const fl4* x4 = (const fl4*)(x + (size_t)tok0 * D_DIM);
    #pragma unroll
    for (int i = 0; i < 16; ++i) {
        int f = tid + i * 256;           // 4096 float4 total
        int r = f >> 5;                  // token row (32 float4 per row)
        int c = (f & 31) * 4;            // d column
        fl4 v = __builtin_nontemporal_load(x4 + f);
        unsigned lo = (unsigned)f2bf(v.x) | ((unsigned)f2bf(v.y) << 16);
        unsigned hi = (unsigned)f2bf(v.z) | ((unsigned)f2bf(v.w) << 16);
        *(uint2*)&sX[r * SA + c] = make_uint2(lo, hi);
    }

    // ---- persistent y accumulators, seeded with b2 (per output column) ----
    v8f accY[2][4];
    #pragma unroll
    for (int tn = 0; tn < 4; ++tn) {
        float bv = b2e[n0 + tn * 16 + lm];
        #pragma unroll
        for (int tm = 0; tm < 2; ++tm)
            #pragma unroll
            for (int r = 0; r < 8; ++r) accY[tm][tn][r] = bv;
    }

    for (int hc = 0; hc < H_DIM; hc += HC) {
        __syncthreads();  // prior GEMM reads of sW1/sW2/sH complete before overwrite

        // ---- stage W1/W2 chunks: b128 global loads, bf16 transposed LDS stores ----
        #pragma unroll
        for (int i = 0; i < 16; ++i) {
            int f  = tid + i * 256;      // 4096 float4 per weight chunk
            int k  = f >> 5;             // 0..127 (row)
            int c4 = (f & 31) * 4;       // column base (coalesced global dim)
            fl4 w1v = ((const fl4*)(W1e + (size_t)k * H_DIM + hc))[f & 31];  // [d][h]
            fl4 w2v = ((const fl4*)(W2e + (size_t)(hc + k) * D_DIM))[f & 31];// [h][d]
            #pragma unroll
            for (int j = 0; j < 4; ++j) {
                sW1[(c4 + j) * SA + k] = f2bf(w1v[j]);   // -> [h][d]
                sW2[(c4 + j) * SA + k] = f2bf(w2v[j]);   // -> [d][h]
            }
        }

        // h accumulators seeded with b1 (per h column)
        v8f accH[2][4];
        #pragma unroll
        for (int tn = 0; tn < 4; ++tn) {
            float bv = b1e[hc + n0 + tn * 16 + lm];
            #pragma unroll
            for (int tm = 0; tm < 2; ++tm)
                #pragma unroll
                for (int r = 0; r < 8; ++r) accH[tm][tn][r] = bv;
        }

        __syncthreads();  // sX (first iter) and weight chunks visible

        // ---- GEMM1: h = x @ W1chunk ----
        #pragma unroll
        for (int kk = 0; kk < D_DIM; kk += 32) {
            v16bf a0 = ldsA(sX, m0 + lm,      kk, half);
            v16bf a1 = ldsA(sX, m0 + 16 + lm, kk, half);
            v16bf bb[4];
            #pragma unroll
            for (int tn = 0; tn < 4; ++tn)
                bb[tn] = ldsB(sW1, n0 + tn * 16 + lm, kk, half);
            #pragma unroll
            for (int tn = 0; tn < 4; ++tn) {
                accH[0][tn] = wmma_bf16(a0, bb[tn], accH[0][tn]);
                accH[1][tn] = wmma_bf16(a1, bb[tn], accH[1][tn]);
            }
        }

        // ---- ReLU, convert to bf16, scatter into sH (C/D layout -> row major) ----
        #pragma unroll
        for (int tm = 0; tm < 2; ++tm)
            #pragma unroll
            for (int tn = 0; tn < 4; ++tn)
                #pragma unroll
                for (int r = 0; r < 8; ++r) {
                    float v = fmaxf(accH[tm][tn][r], 0.0f);
                    sH[(m0 + tm * 16 + r + 8 * half) * SA + n0 + tn * 16 + lm] = f2bf(v);
                }

        __syncthreads();  // full sH tile visible to all waves

        // ---- GEMM2: y += h @ W2chunk ----
        #pragma unroll
        for (int kk = 0; kk < HC; kk += 32) {
            v16bf a0 = ldsA(sH, m0 + lm,      kk, half);
            v16bf a1 = ldsA(sH, m0 + 16 + lm, kk, half);
            v16bf bb[4];
            #pragma unroll
            for (int tn = 0; tn < 4; ++tn)
                bb[tn] = ldsB(sW2, n0 + tn * 16 + lm, kk, half);
            #pragma unroll
            for (int tn = 0; tn < 4; ++tn) {
                accY[0][tn] = wmma_bf16(a0, bb[tn], accY[0][tn]);
                accY[1][tn] = wmma_bf16(a1, bb[tn], accY[1][tn]);
            }
        }
    }

    // ---- epilogue: scale and store fp32 (non-temporal, streaming) ----
    const float scale = __expf(fminf(temperature[0], 4.605170185988091f)); // ln(100)
    float* o = out + (size_t)tok0 * D_DIM;
    #pragma unroll
    for (int tm = 0; tm < 2; ++tm)
        #pragma unroll
        for (int tn = 0; tn < 4; ++tn)
            #pragma unroll
            for (int r = 0; r < 8; ++r)
                __builtin_nontemporal_store(
                    accY[tm][tn][r] * scale,
                    &o[(m0 + tm * 16 + r + 8 * half) * D_DIM + n0 + tn * 16 + lm]);
}

extern "C" void kernel_launch(void* const* d_in, const int* in_sizes, int n_in,
                              void* d_out, int out_size, void* d_ws, size_t ws_size,
                              hipStream_t stream) {
    const float* x    = (const float*)d_in[0];
    const float* W1   = (const float*)d_in[1];
    const float* b1   = (const float*)d_in[2];
    const float* W2   = (const float*)d_in[3];
    const float* b2   = (const float*)d_in[4];
    const float* temp = (const float*)d_in[5];
    float* out = (float*)d_out;

    const int Ntok   = in_sizes[0] / D_DIM;   // 524288
    const int blocks = Ntok / TOK_TILE;       // 4096

    moe_ffn_kernel<<<blocks, 256, 0, stream>>>(x, W1, b1, W2, b2, temp, out);
}
